// GFCN_17222818857324
// MI455X (gfx1250) — matrix operand order repaired and checked by reading, not compiled, not run on hardware
//
#include <hip/hip_runtime.h>
#include <math.h>

typedef _Float16 half_t;
typedef __attribute__((ext_vector_type(16))) _Float16 v16h;
typedef __attribute__((ext_vector_type(8)))  _Float16 v8h;
typedef __attribute__((ext_vector_type(8)))  float    v8f;

// ---------------------------------------------------------------- helpers
__device__ __forceinline__ float eluf(float x) {
    return x > 0.f ? x : (__expf(x) - 1.f);
}

__device__ __forceinline__ void atomicMaxF(float* a, float v) {
    unsigned int* ua = (unsigned int*)a;
    unsigned int cur = __float_as_uint(*a);
    while (__uint_as_float(cur) < v) {
        unsigned int prev = atomicCAS(ua, cur, __float_as_uint(v));
        if (prev == cur) break;
        cur = prev;
    }
}

// SplineConv bilinear basis, K=5 knots per dim (degree 1)
__device__ __forceinline__ void bilinear4(float p0, float p1, int idx[4], float bw[4]) {
    float v0 = p0 * 4.f, v1 = p1 * 4.f;
    float lo0 = fminf(fmaxf(floorf(v0), 0.f), 4.f);
    float lo1 = fminf(fmaxf(floorf(v1), 0.f), 4.f);
    float f0 = v0 - lo0, f1 = v1 - lo1;
    int a0 = (int)lo0, a1 = (int)lo1;
    int b0 = a0 + 1 < 4 ? a0 + 1 : 4;
    int b1 = a1 + 1 < 4 ? a1 + 1 : 4;
    idx[0] = a0 * 5 + a1; idx[1] = a0 * 5 + b1;
    idx[2] = b0 * 5 + a1; idx[3] = b0 * 5 + b1;
    float g0 = 1.f - f0, g1 = 1.f - f1;
    bw[0] = g0 * g1; bw[1] = g0 * f1; bw[2] = f0 * g1; bw[3] = f0 * f1;
}

// ---------------------------------------------------------------- utility kernels
__global__ void fill_k(float* __restrict__ p, float v, int n) {
    int t = blockIdx.x * blockDim.x + threadIdx.x;
    if (t < n) p[t] = v;
}

__global__ void cvt_f32_f16(const float* __restrict__ in, half_t* __restrict__ out, int n) {
    int t = blockIdx.x * blockDim.x + threadIdx.x;
    if (t < n) out[t] = (half_t)in[t];
}

// ---------------------------------------------------------------- segment ops
__global__ void seg_sum(const float* __restrict__ xin, const int* __restrict__ seg,
                        float* __restrict__ sum, float* __restrict__ cnt, int n, int C) {
    int t = blockIdx.x * blockDim.x + threadIdx.x;
    if (t >= n * C) return;
    int i = t / C, c = t - i * C;
    int s = seg[i];
    atomicAdd(&sum[(size_t)s * C + c], xin[t]);
    if (cnt && c == 0) atomicAdd(&cnt[s], 1.f);
}

__global__ void seg_div(float* __restrict__ sum, const float* __restrict__ cnt, int m, int C) {
    int t = blockIdx.x * blockDim.x + threadIdx.x;
    if (t >= m * C) return;
    sum[t] /= fmaxf(cnt[t / C], 1.f);
}

__global__ void seg_max(const float* __restrict__ xin, const int* __restrict__ seg,
                        float* __restrict__ mx, int n, int C) {
    int t = blockIdx.x * blockDim.x + threadIdx.x;
    if (t >= n * C) return;
    int i = t / C, c = t - i * C;
    atomicMaxF(&mx[(size_t)seg[i] * C + c], xin[t]);
}

// w[n,c] = cent[seg[n],c] != 0 ? x[n,c]/cent[seg[n],c] : 0
__global__ void bweights_k(const float* __restrict__ xin, const int* __restrict__ seg,
                           const float* __restrict__ cent, float* __restrict__ w, int n, int C) {
    int t = blockIdx.x * blockDim.x + threadIdx.x;
    if (t >= n * C) return;
    int i = t / C, c = t - i * C;
    float aux = cent[(size_t)seg[i] * C + c];
    w[t] = (aux != 0.f) ? xin[t] / aux : 0.f;
}

// outh[n,c] = bweight(n,c) * xmax[seg[n],c]   (f16, feeds next GEMM A)
__global__ void bweights_mul(const float* __restrict__ xin, const int* __restrict__ seg,
                             const float* __restrict__ cent, const float* __restrict__ xmax,
                             half_t* __restrict__ outh, int n, int C) {
    int t = blockIdx.x * blockDim.x + threadIdx.x;
    if (t >= n * C) return;
    int i = t / C, c = t - i * C;
    int s = seg[i];
    float aux = cent[(size_t)s * C + c];
    float w = (aux != 0.f) ? xin[t] / aux : 0.f;
    outh[t] = (half_t)(w * xmax[(size_t)s * C + c]);
}

// out[n,c] = w[n,c] * h[seg[n],c]
__global__ void gather_mul(const float* __restrict__ w, const float* __restrict__ h,
                           const int* __restrict__ seg, float* __restrict__ out, int n, int C) {
    int t = blockIdx.x * blockDim.x + threadIdx.x;
    if (t >= n * C) return;
    int i = t / C, c = t - i * C;
    out[t] = w[t] * h[(size_t)seg[i] * C + c];
}

// ---------------------------------------------------------------- edge prep (cluster graph)
__global__ void edge_map(const int* __restrict__ ei, const int* __restrict__ cl,
                         const float* __restrict__ pos1, int* __restrict__ e1s,
                         int* __restrict__ e1d, float* __restrict__ dvec,
                         float* __restrict__ valid, float* __restrict__ gmax, int E) {
    int e = blockIdx.x * blockDim.x + threadIdx.x;
    if (e >= E) return;
    int s = cl[ei[e]], d = cl[ei[E + e]];
    e1s[e] = s; e1d[e] = d;
    valid[e] = (s != d) ? 1.f : 0.f;
    float d0 = pos1[2 * d] - pos1[2 * s];
    float d1 = pos1[2 * d + 1] - pos1[2 * s + 1];
    dvec[2 * e] = d0; dvec[2 * e + 1] = d1;
    atomicMaxF(gmax, fmaxf(fabsf(d0), fabsf(d1)));
}

__global__ void edge_attr_fin(float* __restrict__ dvec, const float* __restrict__ gmax, int n2) {
    int t = blockIdx.x * blockDim.x + threadIdx.x;
    if (t >= n2) return;
    float g = *gmax;
    dvec[t] = dvec[t] / (2.f * g) + 0.5f;
}

// ---------------------------------------------------------------- layer 1 (in=1, out=32)
__global__ void edge_l1(const float* __restrict__ x, const float* __restrict__ eattr,
                        const int* __restrict__ ei, const float* __restrict__ W1,
                        float* __restrict__ agg, float* __restrict__ cnt, int E) {
    int e = blockIdx.x * blockDim.x + threadIdx.x;
    if (e >= E) return;
    int s = ei[e], d = ei[E + e];
    int idx[4]; float bw[4];
    bilinear4(eattr[2 * e], eattr[2 * e + 1], idx, bw);
    float xv = x[s];
    float* ap = agg + (size_t)d * 32;
#pragma unroll 8
    for (int o = 0; o < 32; ++o) {
        float w = bw[0] * W1[idx[0] * 32 + o] + bw[1] * W1[idx[1] * 32 + o] +
                  bw[2] * W1[idx[2] * 32 + o] + bw[3] * W1[idx[3] * 32 + o];
        atomicAdd(&ap[o], xv * w);
    }
    atomicAdd(&cnt[d], 1.f);
}

__global__ void node_fin1(float* __restrict__ h, const float* __restrict__ cnt,
                          const float* __restrict__ x, const float* __restrict__ Wr,
                          const float* __restrict__ b, int N) {
    int t = blockIdx.x * blockDim.x + threadIdx.x;
    if (t >= N * 32) return;
    int n = t >> 5, o = t & 31;
    float v = h[t] / fmaxf(cnt[n], 1.f) + x[n] * Wr[o] + b[o];
    h[t] = eluf(v);
}

// ---------------------------------------------------------------- WMMA GEMM path
// Pack B (32 x Ncols or 64 x Ncols, Ncols = 26*Cout: 25*Cout spline cols + Cout root cols
// from Wr) into this file's contiguous-k wave32 WMMA B layout (matches A): [kstep][ntile][lane][j], k = j + 16*(lane>>4).
__global__ void pack_B(const float* __restrict__ W, const float* __restrict__ Wr,
                       half_t* __restrict__ Bpk, int Kd, int Cout) {
    int ntiles = (26 * Cout) >> 4;
    int ksteps = Kd >> 5;
    int total = ksteps * ntiles * 512;
    int t = blockIdx.x * blockDim.x + threadIdx.x;
    if (t >= total) return;
    int j    = t & 15;
    int lane = (t >> 4) & 31;
    int rest = t >> 9;
    int nt = rest % ntiles;
    int ks = rest / ntiles;
    int i = ks * 32 + j + ((lane >> 4) << 4);      // global K row
    int c = (nt << 4) + (lane & 15);               // global column
    float v = 0.f;
    int root = 25 * Cout;
    if (i < Kd) {
        if (c < root) {
            int k = c / Cout, o = c - k * Cout;
            v = W[((size_t)k * Kd + i) * Cout + o];
        } else {
            v = Wr[(size_t)i * Cout + (c - root)];
        }
    }
    Bpk[t] = (half_t)v;
}

// C[M x Ncols] (f16) = A[M x Kd] (f16, row-major) * Bpk.  One 16x16 tile per wave,
// 8 waves per block covering 8 consecutive N tiles. v_wmma_f32_16x16x32_f16.
__global__ void wmma_gemm(const half_t* __restrict__ A, const half_t* __restrict__ Bpk,
                          half_t* __restrict__ C, int M, int Kd, int Ncols) {
    int lane = threadIdx.x & 31;
    int wv   = threadIdx.x >> 5;
    int ntiles = Ncols >> 4;
    int nt = blockIdx.x * 8 + wv;
    int mt = blockIdx.y;
    if (nt >= ntiles) return;
    int g = lane >> 4;
    int mrow = (mt << 4) + (lane & 15);
    int mclamp = mrow < M ? mrow : M - 1;
    const half_t* arow = A + (size_t)mclamp * Kd;
    v8f acc = {0.f, 0.f, 0.f, 0.f, 0.f, 0.f, 0.f, 0.f};
    int ksteps = Kd >> 5;
    for (int ks = 0; ks < ksteps; ++ks) {
        // contiguous A layout (matches pack_B): lanes 0-15 hold K 0..15, lanes 16-31 hold K 16..31
        v16h a = *(const v16h*)(arow + ks * 32 + 16 * g);
        v16h b = *(const v16h*)(Bpk + (((size_t)ks * ntiles + nt) << 9) + (lane << 4));
        acc = __builtin_amdgcn_wmma_f32_16x16x32_f16(false, a, false, b,
                                                     (short)0, acc, false, false);
    }
    half_t* crow = C + (size_t)(mt << 4) * Ncols + (nt << 4) + (lane & 15);
#pragma unroll
    for (int r = 0; r < 8; ++r) {
        int mm = (mt << 4) + r + 8 * g;
        if (mm < M) crow[(size_t)(r + 8 * g) * Ncols] = (half_t)acc[r];
    }
}

// ---------------------------------------------------------------- spline edge gather (layers 2/3)
// One thread per (edge, 16-output chunk): 4 x v16h gathers from XW[src], bilinear combine,
// f32 atomic scatter into agg[dst].
__global__ void edge_spline(const half_t* __restrict__ XW, const float* __restrict__ attr,
                            const float* __restrict__ valid, const int* __restrict__ es,
                            const int* __restrict__ ed, float* __restrict__ agg,
                            float* __restrict__ cnt, int E, int ldn, int Cout) {
    int t = blockIdx.x * blockDim.x + threadIdx.x;
    int segs = Cout >> 4;
    int e = t / segs;
    if (e >= E) return;
    int ob = (t - e * segs) << 4;
    float vld = valid[e];
    int d = ed[e];
    if (ob == 0) atomicAdd(&cnt[d], vld);
    if (vld == 0.f) return;
    int s = es[e];
    int idx[4]; float bw[4];
    bilinear4(attr[2 * e], attr[2 * e + 1], idx, bw);
    const half_t* bas = XW + (size_t)s * ldn + ob;
    float acc[16];
#pragma unroll
    for (int j = 0; j < 16; ++j) acc[j] = 0.f;
#pragma unroll
    for (int k = 0; k < 4; ++k) {
        v16h gv = *(const v16h*)(bas + idx[k] * Cout);
        float w = bw[k];
#pragma unroll
        for (int j = 0; j < 16; ++j) acc[j] += w * (float)gv[j];
    }
    float* ap = agg + (size_t)d * Cout + ob;
#pragma unroll
    for (int j = 0; j < 16; ++j) atomicAdd(&ap[j], acc[j]);
}

// agg -> h: h = elu(agg/max(cnt,1) + XW_root + b)   (root cols appended in GEMM)
__global__ void node_fin23(float* __restrict__ h, const float* __restrict__ cnt,
                           const half_t* __restrict__ XW, int ldn, int rootOff,
                           const float* __restrict__ b, int M, int Cout) {
    int t = blockIdx.x * blockDim.x + threadIdx.x;
    if (t >= M * Cout) return;
    int n = t / Cout, o = t - n * Cout;
    float v = h[t] / fmaxf(cnt[n], 1.f) + (float)XW[(size_t)n * ldn + rootOff + o] + b[o];
    h[t] = eluf(v);
}

// ---------------------------------------------------------------- layer 4 (in=32, out=1) + sigmoid
__global__ void edge_l4(const float* __restrict__ xq, const float* __restrict__ eattr,
                        const int* __restrict__ ei, const float* __restrict__ W4,
                        float* __restrict__ agg, float* __restrict__ cnt, int E) {
    int e = blockIdx.x * blockDim.x + threadIdx.x;
    if (e >= E) return;
    int s = ei[e], d = ei[E + e];
    int idx[4]; float bw[4];
    bilinear4(eattr[2 * e], eattr[2 * e + 1], idx, bw);
    const float* xr = xq + (size_t)s * 32;
    float m = 0.f;
#pragma unroll 8
    for (int i = 0; i < 32; ++i) {
        float w = bw[0] * W4[idx[0] * 32 + i] + bw[1] * W4[idx[1] * 32 + i] +
                  bw[2] * W4[idx[2] * 32 + i] + bw[3] * W4[idx[3] * 32 + i];
        m += xr[i] * w;
    }
    atomicAdd(&agg[d], m);
    atomicAdd(&cnt[d], 1.f);
}

__global__ void node_fin4(const float* __restrict__ agg, const float* __restrict__ cnt,
                          const float* __restrict__ xq, const float* __restrict__ Wr4,
                          const float* __restrict__ b4, float* __restrict__ out, int N) {
    int n = blockIdx.x * blockDim.x + threadIdx.x;
    if (n >= N) return;
    float v = agg[n] / fmaxf(cnt[n], 1.f);
    const float* xr = xq + (size_t)n * 32;
    float r = 0.f;
#pragma unroll 8
    for (int i = 0; i < 32; ++i) r += xr[i] * Wr4[i];
    v += r + b4[0];
    out[n] = 1.f / (1.f + __expf(-v));
}

// ---------------------------------------------------------------- launch
static inline int cdiv(long a, long b) { return (int)((a + b - 1) / b); }

extern "C" void kernel_launch(void* const* d_in, const int* in_sizes, int n_in,
                              void* d_out, int out_size, void* d_ws, size_t ws_size,
                              hipStream_t stream) {
    const float* x     = (const float*)d_in[0];
    const float* pos   = (const float*)d_in[1];
    const float* eattr = (const float*)d_in[2];
    const int*   ei    = (const int*)d_in[3];
    const int*   cl1   = (const int*)d_in[4];
    const int*   cl2   = (const int*)d_in[5];
    const float* W1  = (const float*)d_in[6];
    const float* Wr1 = (const float*)d_in[7];
    const float* b1  = (const float*)d_in[8];
    const float* W2  = (const float*)d_in[9];
    const float* Wr2 = (const float*)d_in[10];
    const float* b2  = (const float*)d_in[11];
    const float* W3  = (const float*)d_in[12];
    const float* Wr3 = (const float*)d_in[13];
    const float* b3  = (const float*)d_in[14];
    const float* W4  = (const float*)d_in[15];
    const float* Wr4 = (const float*)d_in[16];
    const float* b4  = (const float*)d_in[17];

    const int N   = in_sizes[0];       // 100000
    const int E   = in_sizes[3] / 2;   // 800000
    const int Nc1 = in_sizes[5];       // 50000 (cluster2 has N1 entries)
    const int Nc2 = Nc1 / 2;           // 25000

    const int NC2 = 26 * 64;           // 1664 cols (25*64 spline + 64 root)
    const int NC3 = 26 * 32;           // 832 cols
    const int NT2 = NC2 >> 4;          // 104 n-tiles
    const int NT3 = NC3 >> 4;          // 52 n-tiles

    // workspace carve-up
    char* base = (char*)d_ws;
    size_t off = 0;
    auto alloc = [&](size_t bytes) -> void* {
        void* p = base + off;
        off += (bytes + 255) & ~(size_t)255;
        return p;
    };
    float*  h1    = (float*)alloc((size_t)N * 32 * 4);      // agg1 -> h1 in place
    float*  cnt1  = (float*)alloc((size_t)N * 4);
    float*  cent1 = (float*)alloc((size_t)Nc1 * 32 * 4);
    float*  ccnt1 = (float*)alloc((size_t)Nc1 * 4);
    float*  w1s   = (float*)alloc((size_t)N * 32 * 4);
    float*  x1    = (float*)alloc((size_t)Nc1 * 32 * 4);
    half_t* x1h   = (half_t*)alloc((size_t)Nc1 * 32 * 2);
    float*  pos1  = (float*)alloc((size_t)Nc1 * 2 * 4);
    int*    e1s   = (int*)alloc((size_t)E * 4);
    int*    e1d   = (int*)alloc((size_t)E * 4);
    float*  attr1 = (float*)alloc((size_t)E * 2 * 4);       // dvec -> attr in place
    float*  valid1= (float*)alloc((size_t)E * 4);
    float*  gmax  = (float*)alloc(256);
    half_t* B2pk  = (half_t*)alloc((size_t)1 * NT2 * 512 * 2);
    half_t* XW2   = (half_t*)alloc((size_t)Nc1 * NC2 * 2);
    float*  h2    = (float*)alloc((size_t)Nc1 * 64 * 4);    // agg2 -> h2 in place
    float*  cnt2  = (float*)alloc((size_t)Nc1 * 4);
    float*  cent2 = (float*)alloc((size_t)Nc2 * 64 * 4);
    float*  ccnt2 = (float*)alloc((size_t)Nc2 * 4);
    float*  x2    = (float*)alloc((size_t)Nc2 * 64 * 4);
    half_t* xph   = (half_t*)alloc((size_t)Nc1 * 64 * 2);
    half_t* B3pk  = (half_t*)alloc((size_t)2 * NT3 * 512 * 2);
    half_t* XW3   = (half_t*)alloc((size_t)Nc1 * NC3 * 2);
    float*  h3    = (float*)alloc((size_t)Nc1 * 32 * 4);    // agg3 -> h3 in place
    float*  cnt3  = (float*)alloc((size_t)Nc1 * 4);
    float*  xq    = (float*)alloc((size_t)N * 32 * 4);
    float*  agg4  = (float*)alloc((size_t)N * 4);
    float*  cnt4  = (float*)alloc((size_t)N * 4);
    (void)ws_size; (void)n_in; (void)out_size;

    const int TB = 256;

    // ---- layer 1: spline conv on full graph (in=1 -> out=32), then ELU
    hipMemsetAsync(h1, 0, (size_t)N * 32 * 4, stream);
    hipMemsetAsync(cnt1, 0, (size_t)N * 4, stream);
    edge_l1<<<cdiv(E, TB), TB, 0, stream>>>(x, eattr, ei, W1, h1, cnt1, E);
    node_fin1<<<cdiv((long)N * 32, TB), TB, 0, stream>>>(h1, cnt1, x, Wr1, b1, N);

    // ---- pooling level 1: centroids, bweights, segment-max, pos mean, cluster edges
    hipMemsetAsync(cent1, 0, (size_t)Nc1 * 32 * 4, stream);
    hipMemsetAsync(ccnt1, 0, (size_t)Nc1 * 4, stream);
    seg_sum<<<cdiv((long)N * 32, TB), TB, 0, stream>>>(h1, cl1, cent1, ccnt1, N, 32);
    seg_div<<<cdiv((long)Nc1 * 32, TB), TB, 0, stream>>>(cent1, ccnt1, Nc1, 32);
    bweights_k<<<cdiv((long)N * 32, TB), TB, 0, stream>>>(h1, cl1, cent1, w1s, N, 32);
    fill_k<<<cdiv((long)Nc1 * 32, TB), TB, 0, stream>>>(x1, -3.402823466e38f, Nc1 * 32);
    seg_max<<<cdiv((long)N * 32, TB), TB, 0, stream>>>(h1, cl1, x1, N, 32);
    cvt_f32_f16<<<cdiv((long)Nc1 * 32, TB), TB, 0, stream>>>(x1, x1h, Nc1 * 32);
    hipMemsetAsync(pos1, 0, (size_t)Nc1 * 2 * 4, stream);
    seg_sum<<<cdiv((long)N * 2, TB), TB, 0, stream>>>(pos, cl1, pos1, (float*)nullptr, N, 2);
    seg_div<<<cdiv((long)Nc1 * 2, TB), TB, 0, stream>>>(pos1, ccnt1, Nc1, 2);
    hipMemsetAsync(gmax, 0, 4, stream);
    edge_map<<<cdiv(E, TB), TB, 0, stream>>>(ei, cl1, pos1, e1s, e1d, attr1, valid1, gmax, E);
    edge_attr_fin<<<cdiv((long)E * 2, TB), TB, 0, stream>>>(attr1, gmax, E * 2);

    // ---- layer 2: WMMA GEMM XW2 = x1 (N1x32) * [W2|Wr2] (32x1664), spline gather, ELU
    pack_B<<<cdiv((long)NT2 * 512, TB), TB, 0, stream>>>(W2, Wr2, B2pk, 32, 64);
    wmma_gemm<<<dim3(cdiv(NT2, 8), cdiv(Nc1, 16)), TB, 0, stream>>>(x1h, B2pk, XW2, Nc1, 32, NC2);
    hipMemsetAsync(h2, 0, (size_t)Nc1 * 64 * 4, stream);
    hipMemsetAsync(cnt2, 0, (size_t)Nc1 * 4, stream);
    edge_spline<<<cdiv((long)E * 4, TB), TB, 0, stream>>>(XW2, attr1, valid1, e1s, e1d,
                                                          h2, cnt2, E, NC2, 64);
    node_fin23<<<cdiv((long)Nc1 * 64, TB), TB, 0, stream>>>(h2, cnt2, XW2, NC2, 25 * 64, b2, Nc1, 64);

    // ---- pooling level 2: bweights * segment_max gather
    hipMemsetAsync(cent2, 0, (size_t)Nc2 * 64 * 4, stream);
    hipMemsetAsync(ccnt2, 0, (size_t)Nc2 * 4, stream);
    seg_sum<<<cdiv((long)Nc1 * 64, TB), TB, 0, stream>>>(h2, cl2, cent2, ccnt2, Nc1, 64);
    seg_div<<<cdiv((long)Nc2 * 64, TB), TB, 0, stream>>>(cent2, ccnt2, Nc2, 64);
    fill_k<<<cdiv((long)Nc2 * 64, TB), TB, 0, stream>>>(x2, -3.402823466e38f, Nc2 * 64);
    seg_max<<<cdiv((long)Nc1 * 64, TB), TB, 0, stream>>>(h2, cl2, x2, Nc1, 64);
    bweights_mul<<<cdiv((long)Nc1 * 64, TB), TB, 0, stream>>>(h2, cl2, cent2, x2, xph, Nc1, 64);

    // ---- layer 3: WMMA GEMM XW3 = xp (N1x64) * [W3|Wr3] (64x832), spline gather, ELU
    pack_B<<<cdiv((long)2 * NT3 * 512, TB), TB, 0, stream>>>(W3, Wr3, B3pk, 64, 32);
    wmma_gemm<<<dim3(cdiv(NT3, 8), cdiv(Nc1, 16)), TB, 0, stream>>>(xph, B3pk, XW3, Nc1, 64, NC3);
    hipMemsetAsync(h3, 0, (size_t)Nc1 * 32 * 4, stream);
    hipMemsetAsync(cnt3, 0, (size_t)Nc1 * 4, stream);
    edge_spline<<<cdiv((long)E * 2, TB), TB, 0, stream>>>(XW3, attr1, valid1, e1s, e1d,
                                                          h3, cnt3, E, NC3, 32);
    node_fin23<<<cdiv((long)Nc1 * 32, TB), TB, 0, stream>>>(h3, cnt3, XW3, NC3, 25 * 32, b3, Nc1, 32);
    gather_mul<<<cdiv((long)N * 32, TB), TB, 0, stream>>>(w1s, h3, cl1, xq, N, 32);

    // ---- layer 4: spline conv (in=32 -> out=1) + sigmoid
    hipMemsetAsync(agg4, 0, (size_t)N * 4, stream);
    hipMemsetAsync(cnt4, 0, (size_t)N * 4, stream);
    edge_l4<<<cdiv(E, TB), TB, 0, stream>>>(xq, eattr, ei, W4, agg4, cnt4, E);
    node_fin4<<<cdiv(N, TB), TB, 0, stream>>>(agg4, cnt4, xq, Wr4, b4, (float*)d_out, N);
}
